// TransformerBlock_4011499454583
// MI455X (gfx1250) — matrix operand order repaired and checked by reading, not compiled, not run on hardware
//
#include <hip/hip_runtime.h>
#include <hip/hip_bf16.h>

#define E      1024
#define NH     16
#define HD     64
#define SEQ    2048
#define BATCH  2
#define MTOT   (BATCH * SEQ)

typedef __attribute__((ext_vector_type(16))) __bf16 v16bf;
typedef __attribute__((ext_vector_type(8)))  float  v8f;
typedef __attribute__((ext_vector_type(4)))  int    v4i;

#define AS1 __attribute__((address_space(1)))
#define AS3 __attribute__((address_space(3)))

__device__ __forceinline__ float  b2f(__bf16 v) { return (float)v; }
__device__ __forceinline__ __bf16 f2b(float v)  { return (__bf16)v; }

// ---------------------------------------------------------------------------
// Async global->LDS copy (gfx1250 GLOBAL_LOAD_ASYNC_TO_LDS_B128, ASYNCcnt),
// with a safe synchronous fallback if the builtin is not declared.
// Probe-verified signature: (int4 AS1* src, int4 AS3* dst, imm offset, imm cpol)
// ---------------------------------------------------------------------------
#if defined(__gfx1250__) && __has_builtin(__builtin_amdgcn_global_load_async_to_lds_b128)
#define HAVE_ASYNC_LDS 1
#else
#define HAVE_ASYNC_LDS 0
#endif

__device__ __forceinline__ void async_copy16(__bf16* l, const __bf16* g) {
#if HAVE_ASYNC_LDS
  __builtin_amdgcn_global_load_async_to_lds_b128((AS1 v4i*)g, (AS3 v4i*)l, 0, 0);
#else
  *(uint4*)l = *(const uint4*)g;
#endif
}

__device__ __forceinline__ void wait_async_copies() {
#if HAVE_ASYNC_LDS
#if __has_builtin(__builtin_amdgcn_s_wait_asynccnt)
  __builtin_amdgcn_s_wait_asynccnt(0);
#else
  asm volatile("s_wait_asynccnt 0x0" ::: "memory");
#endif
#endif
}

// ---------------------------------------------------------------------------
// Elementwise: f32 -> bf16 conversion (weights)
// ---------------------------------------------------------------------------
__global__ void f32_to_bf16_kernel(const float* __restrict__ src,
                                   __bf16* __restrict__ dst, int n) {
  int i = blockIdx.x * blockDim.x + threadIdx.x;
  if (i < n) dst[i] = f2b(src[i]);
}

// ---------------------------------------------------------------------------
// RoPE on raw x (q = k = rope(x), v = x), emit bf16 copies of both
// ---------------------------------------------------------------------------
__global__ void rope_kernel(const float* __restrict__ x,
                            __bf16* __restrict__ xb,
                            __bf16* __restrict__ rxb) {
  int i = blockIdx.x * blockDim.x + threadIdx.x;  // over MTOT*E
  if (i >= MTOT * E) return;
  int c   = i % E;
  int tok = i / E;
  int pos = tok % SEQ;
  int d   = c % HD;
  int fi  = d & 31;
  float inv = __powf(10000.f, -(float)fi / 32.f);
  float ang = (float)pos * inv;
  float cs = __cosf(ang), sn = __sinf(ang);
  float xv  = x[i];
  float rot = (d < 32) ? -x[i + 32] : x[i - 32];
  xb[i]  = f2b(xv);
  rxb[i] = f2b(xv * cs + rot * sn);
}

// ---------------------------------------------------------------------------
// WMMA GEMM: out[M,N] = A[M,K] @ W[N,K]^T + bias[N] (+ resid[M,N])
// Block = 256 threads (8 waves); block tile 64x128; wave tile 32x32 (2x2 WMMA)
// ---------------------------------------------------------------------------
template <bool OUTF32, bool RESID>
__global__ void gemm_wmma_kernel(const __bf16* __restrict__ A,
                                 const __bf16* __restrict__ W,
                                 const float* __restrict__ bias,
                                 const float* __restrict__ resid,
                                 void* __restrict__ outp,
                                 int M, int N, int K) {
  __shared__ alignas(16) __bf16 As[64 * 32];
  __shared__ alignas(16) __bf16 Ws[128 * 32];
  const int tid  = threadIdx.x;
  const int lane = tid & 31, wave = tid >> 5;
  const int half = lane >> 4, lm = lane & 15;
  const int waveM = wave & 1, waveN = wave >> 1;
  const int gm0 = blockIdx.x * 64, gn0 = blockIdx.y * 128;

  v8f acc[2][2] = {};

  for (int kk = 0; kk < K; kk += 32) {
    {  // stage A tile 64x32 (8 bf16 per thread, async 16B copy into LDS)
      int idx = tid * 8;
      int r = idx >> 5, c = idx & 31;
      async_copy16(&As[idx], &A[(size_t)(gm0 + r) * K + kk + c]);
    }
    for (int t = 0; t < 2; ++t) {  // stage W tile 128x32
      int idx = (tid + t * 256) * 8;
      int r = idx >> 5, c = idx & 31;
      async_copy16(&Ws[idx], &W[(size_t)(gn0 + r) * K + kk + c]);
    }
    if (kk + 32 < K) {  // speculative prefetch of next A tile -> global_prefetch_b8
      __builtin_prefetch(&A[(size_t)(gm0 + (tid >> 2)) * K + kk + 32], 0, 1);
    }
    wait_async_copies();
    __syncthreads();

    // A fragment: lane = M, K pattern {0..7,16..23} / {8..15,24..31} by lane half
    v16bf a[2], b[2];
    for (int mi2 = 0; mi2 < 2; ++mi2) {
      int row = waveM * 32 + mi2 * 16 + lm;
      for (int j = 0; j < 8; ++j) {
        int kq = ((j < 4) ? (2 * j) : (16 + 2 * (j - 4))) + half * 8;
        a[mi2][2 * j]     = As[row * 32 + kq];
        a[mi2][2 * j + 1] = As[row * 32 + kq + 1];
      }
    }
    // B fragment: lane = N, K pattern {0..7,16..23} / {8..15,24..31} by lane half, 2 packed per VGPR
    for (int ni = 0; ni < 2; ++ni) {
      int nrow = waveN * 32 + ni * 16 + lm;
      for (int j = 0; j < 8; ++j) {
        int kq = ((j < 4) ? (2 * j) : (16 + 2 * (j - 4))) + half * 8;
        b[ni][2 * j]     = Ws[nrow * 32 + kq];
        b[ni][2 * j + 1] = Ws[nrow * 32 + kq + 1];
      }
    }
    for (int mi2 = 0; mi2 < 2; ++mi2)
      for (int ni = 0; ni < 2; ++ni)
        acc[mi2][ni] = __builtin_amdgcn_wmma_f32_16x16x32_bf16(
            false, a[mi2], false, b[ni], (short)0, acc[mi2][ni], false, false);
    __syncthreads();
  }

  // Epilogue: C layout (VGPR r -> M = r + 8*half, N = lane&15)
  for (int mi2 = 0; mi2 < 2; ++mi2)
    for (int ni = 0; ni < 2; ++ni)
      for (int r = 0; r < 8; ++r) {
        int row = gm0 + waveM * 32 + mi2 * 16 + half * 8 + r;
        int col = gn0 + waveN * 32 + ni * 16 + lm;
        float v = acc[mi2][ni][r] + bias[col];
        if (RESID) v += resid[(size_t)row * N + col];
        if (OUTF32) ((float*)outp)[(size_t)row * N + col] = v;
        else        ((__bf16*)outp)[(size_t)row * N + col] = f2b(v);
      }
}

// ---------------------------------------------------------------------------
// Flash attention: grid (B*H, SEQ/64), block 128 (4 waves); wave = 16 q rows.
// Online softmax; S never hits HBM. P transposed via per-wave LDS region.
// ---------------------------------------------------------------------------
__global__ void flash_attn_kernel(const __bf16* __restrict__ qp,
                                  const __bf16* __restrict__ kp,
                                  const __bf16* __restrict__ vp,
                                  __bf16* __restrict__ ctx) {
  __shared__ alignas(16) __bf16 Ks[32 * 64];
  __shared__ alignas(16) __bf16 Vs[32 * 64];
  __shared__ alignas(16) __bf16 Ps[4 * 16 * 32];
  const int tid  = threadIdx.x;
  const int lane = tid & 31, wave = tid >> 5;
  const int half = lane >> 4, lm = lane & 15;
  const int bh = blockIdx.x;
  const int b = bh / NH, h = bh % NH;
  const int q0 = blockIdx.y * 64 + wave * 16;
  const size_t rowbase = (size_t)b * SEQ;

  // Q fragments for K-dim (head dim) steps 0..31 and 32..63, pre-scaled 1/sqrt(D)
  v16bf qa[2];
  for (int ks = 0; ks < 2; ++ks) {
    const __bf16* qrow = qp + (rowbase + q0 + lm) * E + h * HD;
    for (int j = 0; j < 8; ++j) {
      int kq = ks * 32 + ((j < 4) ? (2 * j) : (16 + 2 * (j - 4))) + half * 8;
      qa[ks][2 * j]     = f2b(b2f(qrow[kq])     * 0.125f);
      qa[ks][2 * j + 1] = f2b(b2f(qrow[kq + 1]) * 0.125f);
    }
  }

  float mrun[8], lrun[8];
  for (int r = 0; r < 8; ++r) { mrun[r] = -1e30f; lrun[r] = 0.f; }
  v8f o[4] = {};
  __bf16* Pw = &Ps[wave * 16 * 32];

  for (int kt = 0; kt < SEQ; kt += 32) {
    // cooperative stage of 32 keys x 64 dims of K and V (async to LDS)
    for (int t = 0; t < 2; ++t) {
      int idx = (tid + t * 128) * 8;
      int r = idx >> 6, c = idx & 63;
      async_copy16(&Ks[idx], &kp[(rowbase + kt + r) * E + h * HD + c]);
      async_copy16(&Vs[idx], &vp[(rowbase + kt + r) * E + h * HD + c]);
    }
    wait_async_copies();
    __syncthreads();

    // S tiles: keys [kt, kt+16) and [kt+16, kt+32)
    v8f s0 = {}, s1 = {};
    for (int ks = 0; ks < 2; ++ks) {
      v16bf bk0, bk1;
      for (int j = 0; j < 8; ++j) {
        int dq = ks * 32 + ((j < 4) ? (2 * j) : (16 + 2 * (j - 4))) + half * 8;
        bk0[2 * j]     = Ks[lm * 64 + dq];
        bk0[2 * j + 1] = Ks[lm * 64 + dq + 1];
        bk1[2 * j]     = Ks[(16 + lm) * 64 + dq];
        bk1[2 * j + 1] = Ks[(16 + lm) * 64 + dq + 1];
      }
      s0 = __builtin_amdgcn_wmma_f32_16x16x32_bf16(false, qa[ks], false, bk0,
                                                   (short)0, s0, false, false);
      s1 = __builtin_amdgcn_wmma_f32_16x16x32_bf16(false, qa[ks], false, bk1,
                                                   (short)0, s1, false, false);
    }

    // online softmax per row (row = r + 8*half; 16 cols live in 16 lanes)
    float pr0[8], pr1[8];
    for (int r = 0; r < 8; ++r) {
      float v = fmaxf(s0[r], s1[r]);
      for (int off = 8; off >= 1; off >>= 1)
        v = fmaxf(v, __shfl_xor(v, off, 32));
      float mnew  = fmaxf(mrun[r], v);
      float scale = __expf(mrun[r] - mnew);
      float p0 = __expf(s0[r] - mnew);
      float p1 = __expf(s1[r] - mnew);
      float rs = p0 + p1;
      for (int off = 8; off >= 1; off >>= 1)
        rs += __shfl_xor(rs, off, 32);
      lrun[r] = lrun[r] * scale + rs;
      mrun[r] = mnew;
      for (int t = 0; t < 4; ++t) o[t][r] *= scale;
      pr0[r] = p0; pr1[r] = p1;
    }

    // stage P (C layout -> row-major) in this wave's LDS region
    for (int r = 0; r < 8; ++r) {
      Pw[(half * 8 + r) * 32 + lm]      = f2b(pr0[r]);
      Pw[(half * 8 + r) * 32 + 16 + lm] = f2b(pr1[r]);
    }
    __syncthreads();

    // re-load P as A fragment, accumulate O += P @ V
    v16bf pa;
    for (int j = 0; j < 8; ++j) {
      int kq = ((j < 4) ? (2 * j) : (16 + 2 * (j - 4))) + half * 8;
      pa[2 * j]     = Pw[lm * 32 + kq];
      pa[2 * j + 1] = Pw[lm * 32 + kq + 1];
    }
    for (int t = 0; t < 4; ++t) {
      v16bf vb;
      for (int j = 0; j < 8; ++j) {
        int krow = ((j < 4) ? (2 * j) : (16 + 2 * (j - 4))) + half * 8;
        vb[2 * j]     = Vs[krow * 64 + t * 16 + lm];
        vb[2 * j + 1] = Vs[(krow + 1) * 64 + t * 16 + lm];
      }
      o[t] = __builtin_amdgcn_wmma_f32_16x16x32_bf16(false, pa, false, vb,
                                                     (short)0, o[t], false, false);
    }
    __syncthreads();
  }

  for (int t = 0; t < 4; ++t)
    for (int r = 0; r < 8; ++r) {
      int row = q0 + half * 8 + r;
      int col = h * HD + t * 16 + lm;
      ctx[(rowbase + row) * E + col] = f2b(o[t][r] / lrun[r]);
    }
}

// ---------------------------------------------------------------------------
// LayerNorm over E=1024; optional bf16 mirror output
// ---------------------------------------------------------------------------
__global__ void ln_kernel(const float* __restrict__ inp,
                          const float* __restrict__ g, const float* __restrict__ bta,
                          float* __restrict__ out, __bf16* __restrict__ outb) {
  __shared__ float sh[E];
  __shared__ float red[256];
  int row = blockIdx.x;
  const float* xr = inp + (size_t)row * E;
  float s = 0.f;
  for (int c = threadIdx.x; c < E; c += 256) { float v = xr[c]; sh[c] = v; s += v; }
  red[threadIdx.x] = s; __syncthreads();
  for (int st = 128; st > 0; st >>= 1) {
    if (threadIdx.x < st) red[threadIdx.x] += red[threadIdx.x + st];
    __syncthreads();
  }
  float mean = red[0] / E;
  __syncthreads();
  float s2 = 0.f;
  for (int c = threadIdx.x; c < E; c += 256) { float d = sh[c] - mean; s2 += d * d; }
  red[threadIdx.x] = s2; __syncthreads();
  for (int st = 128; st > 0; st >>= 1) {
    if (threadIdx.x < st) red[threadIdx.x] += red[threadIdx.x + st];
    __syncthreads();
  }
  float rstd = rsqrtf(red[0] / E + 1e-5f);
  for (int c = threadIdx.x; c < E; c += 256) {
    float v = (sh[c] - mean) * rstd * g[c] + bta[c];
    out[(size_t)row * E + c] = v;
    if (outb) outb[(size_t)row * E + c] = f2b(v);
  }
}

// ---------------------------------------------------------------------------
// GeGLU gate (exact erf gelu) + residual + LN2 -> final f32 output
// ---------------------------------------------------------------------------
__global__ void geglu_ln2_kernel(const float* __restrict__ proj,
                                 const float* __restrict__ h,
                                 const float* __restrict__ g, const float* __restrict__ bta,
                                 float* __restrict__ out) {
  __shared__ float sh[E];
  __shared__ float red[256];
  int row = blockIdx.x;
  const float* pr = proj + (size_t)row * 2 * E;
  const float* hr = h + (size_t)row * E;
  float s = 0.f;
  for (int c = threadIdx.x; c < E; c += 256) {
    float val = pr[c], gate = pr[c + E];
    float ge = 0.5f * gate * (1.f + erff(gate * 0.70710678118f));
    float t = hr[c] + val * ge;
    sh[c] = t; s += t;
  }
  red[threadIdx.x] = s; __syncthreads();
  for (int st = 128; st > 0; st >>= 1) {
    if (threadIdx.x < st) red[threadIdx.x] += red[threadIdx.x + st];
    __syncthreads();
  }
  float mean = red[0] / E;
  __syncthreads();
  float s2 = 0.f;
  for (int c = threadIdx.x; c < E; c += 256) { float d = sh[c] - mean; s2 += d * d; }
  red[threadIdx.x] = s2; __syncthreads();
  for (int st = 128; st > 0; st >>= 1) {
    if (threadIdx.x < st) red[threadIdx.x] += red[threadIdx.x + st];
    __syncthreads();
  }
  float rstd = rsqrtf(red[0] / E + 1e-5f);
  for (int c = threadIdx.x; c < E; c += 256)
    out[(size_t)row * E + c] = (sh[c] - mean) * rstd * g[c] + bta[c];
}

// ---------------------------------------------------------------------------
// Host-side orchestration
// ---------------------------------------------------------------------------
extern "C" void kernel_launch(void* const* d_in, const int* in_sizes, int n_in,
                              void* d_out, int out_size, void* d_ws, size_t ws_size,
                              hipStream_t stream) {
  (void)in_sizes; (void)n_in; (void)out_size; (void)ws_size;
  const float* x          = (const float*)d_in[0];
  const float* in_proj_w  = (const float*)d_in[1];
  const float* in_proj_b  = (const float*)d_in[2];
  const float* out_proj_w = (const float*)d_in[3];
  const float* out_proj_b = (const float*)d_in[4];
  const float* geglu_w    = (const float*)d_in[5];
  const float* geglu_b    = (const float*)d_in[6];
  const float* ln1_g = (const float*)d_in[7];
  const float* ln1_b = (const float*)d_in[8];
  const float* ln2_g = (const float*)d_in[9];
  const float* ln2_b = (const float*)d_in[10];
  float* out = (float*)d_out;

  // workspace carve-up (bytes)
  char* ws = (char*)d_ws;
  size_t off = 0;
  auto carve = [&](size_t bytes) { char* p = ws + off; off += (bytes + 255) & ~(size_t)255; return p; };
  __bf16* xb   = (__bf16*)carve((size_t)MTOT * E * 2);
  __bf16* rxb  = (__bf16*)carve((size_t)MTOT * E * 2);
  __bf16* wqkv = (__bf16*)carve((size_t)3 * E * E * 2);
  __bf16* wo   = (__bf16*)carve((size_t)E * E * 2);
  __bf16* wg   = (__bf16*)carve((size_t)2 * E * E * 2);
  __bf16* qpb  = (__bf16*)carve((size_t)MTOT * E * 2);
  __bf16* kpb  = (__bf16*)carve((size_t)MTOT * E * 2);
  __bf16* vpb  = (__bf16*)carve((size_t)MTOT * E * 2);
  __bf16* ctxb = (__bf16*)carve((size_t)MTOT * E * 2);
  float*  h1   = (float*)carve((size_t)MTOT * E * 4);
  float*  h    = (float*)carve((size_t)MTOT * E * 4);
  __bf16* hb   = (__bf16*)carve((size_t)MTOT * E * 2);
  float*  proj = (float*)carve((size_t)MTOT * 2 * E * 4);

  // 1) weights -> bf16
  f32_to_bf16_kernel<<<(3 * E * E + 255) / 256, 256, 0, stream>>>(in_proj_w, wqkv, 3 * E * E);
  f32_to_bf16_kernel<<<(E * E + 255) / 256, 256, 0, stream>>>(out_proj_w, wo, E * E);
  f32_to_bf16_kernel<<<(2 * E * E + 255) / 256, 256, 0, stream>>>(geglu_w, wg, 2 * E * E);

  // 2) RoPE + x -> bf16
  rope_kernel<<<(MTOT * E + 255) / 256, 256, 0, stream>>>(x, xb, rxb);

  // 3-5) in-proj GEMMs: qp/kp from rope(x), vp from x
  dim3 ggrid(MTOT / 64, E / 128);
  gemm_wmma_kernel<false, false><<<ggrid, 256, 0, stream>>>(
      rxb, wqkv,             in_proj_b,         nullptr, qpb, MTOT, E, E);
  gemm_wmma_kernel<false, false><<<ggrid, 256, 0, stream>>>(
      rxb, wqkv + E * E,     in_proj_b + E,     nullptr, kpb, MTOT, E, E);
  gemm_wmma_kernel<false, false><<<ggrid, 256, 0, stream>>>(
      xb,  wqkv + 2 * E * E, in_proj_b + 2 * E, nullptr, vpb, MTOT, E, E);

  // 6) flash attention
  flash_attn_kernel<<<dim3(BATCH * NH, SEQ / 64), 128, 0, stream>>>(qpb, kpb, vpb, ctxb);

  // 7) out-proj + residual (f32)
  gemm_wmma_kernel<true, true><<<ggrid, 256, 0, stream>>>(
      ctxb, wo, out_proj_b, x, h1, MTOT, E, E);

  // 8) LN1 -> h (f32) and hb (bf16)
  ln_kernel<<<MTOT, 256, 0, stream>>>(h1, ln1_g, ln1_b, h, hb);

  // 9) GeGLU projection (N = 2E, f32 out)
  gemm_wmma_kernel<true, false><<<dim3(MTOT / 64, 2 * E / 128), 256, 0, stream>>>(
      hb, wg, geglu_b, nullptr, proj, MTOT, 2 * E, E);

  // 10) gate + residual + LN2 -> d_out
  geglu_ln2_kernel<<<MTOT, 256, 0, stream>>>(proj, h, ln2_g, ln2_b, out);
}
